// BiMambaBlock_3951369912773
// MI455X (gfx1250) — compile-verified, matrix operand order checked by reading, not run on hardware
//
#include <hip/hip_runtime.h>

// ---------------------------------------------------------------------------
// BiMamba block for MI455X (gfx1250, wave32, WMMA).
// GEMMs: v_wmma_f32_16x16x32_bf16, 32x32 wave tiles (4 WMMA / 8 b128 loads).
// Scan: 32-way chunked linear-recurrence scan with async-to-LDS B/C broadcast.
// ---------------------------------------------------------------------------

typedef __attribute__((ext_vector_type(16))) __bf16 v16bf;
typedef __attribute__((ext_vector_type(8)))  __bf16 v8bf;
typedef __attribute__((ext_vector_type(8)))  float  v8f;

static constexpr int B_  = 2;
static constexpr int L_  = 2048;
static constexpr int D_  = 768;
static constexpr int DI_ = 1536;           // E*D
static constexpr int S_  = 16;
static constexpr int R_  = 48;
static constexpr int NP_ = R_ + 2 * S_;    // 80
static constexpr int N1_ = 2 * DI_;        // 3072
static constexpr int D2_ = 2 * D_;         // 1536
static constexpr int M_  = B_ * L_;        // 4096
static constexpr int T_  = 32;             // scan chunks
static constexpr int LC_ = L_ / T_;        // 64 steps per chunk

// ---------------- small device helpers ----------------

__device__ __forceinline__ __bf16 f2bf(float f) {
  unsigned u = __builtin_bit_cast(unsigned, f);
  unsigned r = (u + 0x7FFFu + ((u >> 16) & 1u)) >> 16;   // round-nearest-even
  unsigned short s = (unsigned short)r;
  return __builtin_bit_cast(__bf16, s);
}

__device__ __forceinline__ float siluf(float x) {
  return x / (1.0f + __expf(-x));
}

__device__ __forceinline__ float softplusf(float x) {
  return (x > 20.0f) ? x : log1pf(__expf(x));
}

__device__ __forceinline__ int revrow(int r) {           // reverse l within batch
  return (r & ~(L_ - 1)) | ((L_ - 1) - (r & (L_ - 1)));
}

__device__ __forceinline__ size_t pf_idx(int b, int j, int s, int c) {
  return (((size_t)(b * T_ + j) * S_ + s) * DI_) + c;
}

// one float per lane: global -> LDS async copy (CDNA5 async engine, ASYNCcnt)
__device__ __forceinline__ void async_f32_to_lds(const float* gsrc, float* ldst) {
  unsigned lds = (unsigned)(uintptr_t)ldst;
  asm volatile("global_load_async_to_lds_b32 %0, %1, off"
               :: "v"(lds), "v"(gsrc) : "memory");
}

__device__ __forceinline__ void wait_asynccnt0() {
  asm volatile("s_wait_asynccnt 0x0" ::: "memory");
}

// Fragment load matching CDNA5 16-bit A/B WMMA layout:
// lane h=0 (lanes 0-15):  elems 0..7 -> K k0+0..7,  elems 8..15 -> K k0+16..23
// lane h=1 (lanes 16-31): elems 0..7 -> K k0+8..15, elems 8..15 -> K k0+24..31
__device__ __forceinline__ v16bf frag_load_fast(const __bf16* rowp, int k0, int h) {
  v8bf lo = *(const v8bf*)(rowp + k0 + h * 8);
  v8bf hi = *(const v8bf*)(rowp + k0 + 16 + h * 8);
  return __builtin_shufflevector(lo, hi, 0, 1, 2, 3, 4, 5, 6, 7,
                                 8, 9, 10, 11, 12, 13, 14, 15);
}

__device__ __forceinline__ v16bf frag_load_tail(const __bf16* rowp, int k0, int h, int Kd) {
  v16bf a;
#pragma unroll
  for (int i = 0; i < 16; ++i) {
    int k = k0 + ((i < 8) ? (h * 8 + i) : (16 + h * 8 + (i - 8)));
    unsigned short u = 0;
    if (k < Kd) u = __builtin_bit_cast(unsigned short, rowp[k]);
    a[i] = __builtin_bit_cast(__bf16, u);
  }
  return a;
}

// ---------------- f32 -> bf16 convert ----------------

__global__ __launch_bounds__(256) void cvt_bf16(const float* __restrict__ src,
                                                __bf16* __restrict__ dst, int n) {
  int i = blockIdx.x * 256 + threadIdx.x;
  if (i < n) dst[i] = f2bf(src[i]);
}

// ---------------- generic WMMA GEMM: C[M x N] = A[M x Kd] * W[N x Kd]^T -----
// 256 threads = 8 waves; block tile 64(M) x 128(N); wave tile 32(M) x 32(N):
// 4 accumulators, 4 WMMA per 8 b128 loads per K-step of 32.
// epilogue modes: 0 f32 | 1 f32+bf16 | 2 bias+softplus f32 | 3 bf16 | 4 bias f32

__global__ __launch_bounds__(256) void gemm_bf16_wmma(
    const __bf16* __restrict__ A, int ldA,
    const __bf16* __restrict__ W, int ldW,
    int N, int Kd,
    float* __restrict__ outF, __bf16* __restrict__ outB, int ldC,
    const float* __restrict__ bias,
    int mode, int revIn, int revOut) {
  const int tid  = threadIdx.x;
  const int lane = tid & 31;
  const int wave = tid >> 5;
  const int wm = wave >> 2, wn = wave & 3;
  const int m0 = blockIdx.y * 64 + wm * 32;
  const int n0 = blockIdx.x * 128 + wn * 32;
  if (n0 >= N) return;                       // wave-uniform (N % 16 == 0)
  const bool c1ok = (n0 + 16) < N;           // second 16-col subtile valid?
  const int h = lane >> 4, idx = lane & 15;

  int ra0 = m0 + idx, ra1 = m0 + 16 + idx;
  if (revIn) { ra0 = revrow(ra0); ra1 = revrow(ra1); }
  const __bf16* pA0 = A + (size_t)ra0 * ldA;
  const __bf16* pA1 = A + (size_t)ra1 * ldA;
  const __bf16* pB0 = W + (size_t)(n0 + idx) * ldW;
  const int rb1 = c1ok ? (n0 + 16 + idx) : (n0 + idx);   // clamp for safe loads
  const __bf16* pB1 = W + (size_t)rb1 * ldW;

  v8f c00 = {}, c01 = {}, c10 = {}, c11 = {};
  const int kfull = Kd & ~31;
  for (int k0 = 0; k0 < kfull; k0 += 32) {
    __builtin_prefetch(pA0 + k0 + 128, 0, 3);   // global_prefetch_b8 (near)
    __builtin_prefetch(pB0 + k0 + 128, 0, 3);
    v16bf a0 = frag_load_fast(pA0, k0, h);
    v16bf a1 = frag_load_fast(pA1, k0, h);
    v16bf b0 = frag_load_fast(pB0, k0, h);
    v16bf b1 = frag_load_fast(pB1, k0, h);
    c00 = __builtin_amdgcn_wmma_f32_16x16x32_bf16(false, a0, false, b0,
                                                  (short)0, c00, false, false);
    c10 = __builtin_amdgcn_wmma_f32_16x16x32_bf16(false, a1, false, b0,
                                                  (short)0, c10, false, false);
    c01 = __builtin_amdgcn_wmma_f32_16x16x32_bf16(false, a0, false, b1,
                                                  (short)0, c01, false, false);
    c11 = __builtin_amdgcn_wmma_f32_16x16x32_bf16(false, a1, false, b1,
                                                  (short)0, c11, false, false);
  }
  if (kfull < Kd) {                           // K tail (dt GEMM: Kd = 48)
    v16bf a0 = frag_load_tail(pA0, kfull, h, Kd);
    v16bf a1 = frag_load_tail(pA1, kfull, h, Kd);
    v16bf b0 = frag_load_tail(pB0, kfull, h, Kd);
    v16bf b1 = frag_load_tail(pB1, kfull, h, Kd);
    c00 = __builtin_amdgcn_wmma_f32_16x16x32_bf16(false, a0, false, b0,
                                                  (short)0, c00, false, false);
    c10 = __builtin_amdgcn_wmma_f32_16x16x32_bf16(false, a1, false, b0,
                                                  (short)0, c10, false, false);
    c01 = __builtin_amdgcn_wmma_f32_16x16x32_bf16(false, a0, false, b1,
                                                  (short)0, c01, false, false);
    c11 = __builtin_amdgcn_wmma_f32_16x16x32_bf16(false, a1, false, b1,
                                                  (short)0, c11, false, false);
  }

  v8f accs[2][2] = {{c00, c10}, {c01, c11}};
#pragma unroll
  for (int g = 0; g < 2; ++g) {
    if (g == 1 && !c1ok) break;
    const int col = n0 + g * 16 + idx;
    const float bv = (mode == 2 || mode == 4) ? bias[col] : 0.0f;
#pragma unroll
    for (int part = 0; part < 2; ++part) {
      v8f acc = accs[g][part];
#pragma unroll
      for (int r = 0; r < 8; ++r) {
        int mrow = m0 + part * 16 + h * 8 + r;  // C layout: VGPR r -> M = 8*h + r
        int drow = revOut ? revrow(mrow) : mrow;
        float v = acc[r];
        size_t o = (size_t)drow * ldC + col;
        if (mode == 0)      { outF[o] = v; }
        else if (mode == 1) { outF[o] = v; outB[o] = f2bf(v); }
        else if (mode == 2) { outF[o] = softplusf(v + bv); }
        else if (mode == 3) { outB[o] = f2bf(v); }
        else                { outF[o] = v + bv; }
      }
    }
  }
}

// ---------------- depthwise causal conv (K=4) + bias + SiLU ----------------

__global__ __launch_bounds__(256) void conv_silu(
    const float* __restrict__ xz, const float* __restrict__ cw,
    const float* __restrict__ cb, float* __restrict__ xa,
    __bf16* __restrict__ xab) {
  int e = blockIdx.x * 256 + threadIdx.x;     // exact grid: M_*DI_ / 256
  int m = e / DI_;
  int c = e - m * DI_;
  int l = m & (L_ - 1);
  float v = cb[c];
#pragma unroll
  for (int j = 0; j < 4; ++j) {
    int t = l - 3 + j;
    if (t >= 0) v += cw[c * 4 + j] * xz[(size_t)(m - l + t) * N1_ + c];
  }
  v = siluf(v);
  xa[e]  = v;
  xab[e] = f2bf(v);
}

// ---------------- chunked selective scan ----------------
// h_t = a_t*h_{t-1} + u_t  per (b,c,s);  a_t = exp(dt*A),  u_t = dt*B_t*x_t.
// pass1: per chunk, from h=0: store (prod a, final h).  pass2: stitch chunk
// boundary states serially (tiny).  pass3: recompute with corrected h_init,
// fusing y = sum_s h*C + Dp*x, the silu(z) gate and bf16 conversion.
// B_t/C_t (32 uniform floats) are double-buffered in LDS via async-to-LDS.

__global__ __launch_bounds__(256) void scan_part1(
    const float* __restrict__ xa, const float* __restrict__ dt,
    const float* __restrict__ dbc, const float* __restrict__ A_log,
    float* __restrict__ P, float* __restrict__ F) {
  const int tid = threadIdx.x;
  const int lane = tid & 31;
  const int wave = tid >> 5;
  const int c = blockIdx.x * 256 + tid;
  const int j = blockIdx.y;
  const int b = blockIdx.z;

  __shared__ float BC[2][32];

  float As[S_];
#pragma unroll
  for (int s = 0; s < S_; ++s) As[s] = -__expf(A_log[c * S_ + s]);
  float hreg[S_], pr[S_];
#pragma unroll
  for (int s = 0; s < S_; ++s) { hreg[s] = 0.0f; pr[s] = 1.0f; }

  const int row0 = b * L_ + j * LC_;
  if (wave == 0) {
    async_f32_to_lds(dbc + (size_t)row0 * NP_ + R_ + lane, &BC[0][lane]);
    wait_asynccnt0();
  }
  __syncthreads();

  for (int tt = 0; tt < LC_; ++tt) {
    const int cur = tt & 1;
    const int row = row0 + tt;
    if (wave == 0 && tt + 1 < LC_)
      async_f32_to_lds(dbc + (size_t)(row + 1) * NP_ + R_ + lane, &BC[cur ^ 1][lane]);
    const float xt  = xa[(size_t)row * DI_ + c];
    const float dtt = dt[(size_t)row * DI_ + c];
#pragma unroll
    for (int s = 0; s < S_; ++s) {
      float dA = __expf(dtt * As[s]);
      pr[s] *= dA;
      hreg[s] = dA * hreg[s] + dtt * BC[cur][s] * xt;
    }
    if (wave == 0) wait_asynccnt0();
    __syncthreads();
  }
#pragma unroll
  for (int s = 0; s < S_; ++s) {
    P[pf_idx(b, j, s, c)] = pr[s];
    F[pf_idx(b, j, s, c)] = hreg[s];
  }
}

__global__ __launch_bounds__(256) void scan_fix(
    const float* __restrict__ P, const float* __restrict__ F,
    float* __restrict__ Hinit) {
  const int e = blockIdx.x * 256 + threadIdx.x;  // exact: B_*S_*DI_ / 256
  const int c = e % DI_;
  const int s = (e / DI_) % S_;
  const int b = e / (DI_ * S_);
  float hp = 0.0f;
  for (int j = 0; j < T_; ++j) {
    const size_t id = pf_idx(b, j, s, c);
    Hinit[id] = hp;
    hp = P[id] * hp + F[id];
  }
}

__global__ __launch_bounds__(256) void scan_part2(
    const float* __restrict__ xa, const float* __restrict__ dt,
    const float* __restrict__ dbc, const float* __restrict__ A_log,
    const float* __restrict__ Dp, const float* __restrict__ Hinit,
    const float* __restrict__ xz, __bf16* __restrict__ yg) {
  const int tid = threadIdx.x;
  const int lane = tid & 31;
  const int wave = tid >> 5;
  const int c = blockIdx.x * 256 + tid;
  const int j = blockIdx.y;
  const int b = blockIdx.z;

  __shared__ float BC[2][32];

  float As[S_];
#pragma unroll
  for (int s = 0; s < S_; ++s) As[s] = -__expf(A_log[c * S_ + s]);
  float hreg[S_];
#pragma unroll
  for (int s = 0; s < S_; ++s) hreg[s] = Hinit[pf_idx(b, j, s, c)];
  const float dpc = Dp[c];

  const int row0 = b * L_ + j * LC_;
  if (wave == 0) {
    async_f32_to_lds(dbc + (size_t)row0 * NP_ + R_ + lane, &BC[0][lane]);
    wait_asynccnt0();
  }
  __syncthreads();

  for (int tt = 0; tt < LC_; ++tt) {
    const int cur = tt & 1;
    const int row = row0 + tt;
    if (wave == 0 && tt + 1 < LC_)
      async_f32_to_lds(dbc + (size_t)(row + 1) * NP_ + R_ + lane, &BC[cur ^ 1][lane]);
    const float xt  = xa[(size_t)row * DI_ + c];
    const float dtt = dt[(size_t)row * DI_ + c];
    float accv = 0.0f;
#pragma unroll
    for (int s = 0; s < S_; ++s) {
      float dA = __expf(dtt * As[s]);
      hreg[s] = dA * hreg[s] + dtt * BC[cur][s] * xt;
      accv += hreg[s] * BC[cur][16 + s];
    }
    const float yv = accv + dpc * xt;
    const float z  = xz[(size_t)row * N1_ + DI_ + c];
    yg[(size_t)row * DI_ + c] = f2bf(yv * siluf(z));
    if (wave == 0) wait_asynccnt0();
    __syncthreads();
  }
}

// ---------------- residual + LayerNorm ----------------

__global__ __launch_bounds__(256) void ln_k(
    const float* __restrict__ x, const float* __restrict__ pj,
    const float* __restrict__ g, const float* __restrict__ bb,
    float* __restrict__ out) {
  const int m = blockIdx.x;
  const int tid = threadIdx.x;
  __shared__ float s1[256];
  __shared__ float s2[256];
  float vals[3];
  float a0 = 0.0f, a1 = 0.0f;
#pragma unroll
  for (int i = 0; i < 3; ++i) {
    int d = tid + i * 256;
    float v = x[(size_t)m * D_ + d] + pj[(size_t)m * D_ + d];
    vals[i] = v;
    a0 += v;
    a1 += v * v;
  }
  s1[tid] = a0;
  s2[tid] = a1;
  __syncthreads();
  for (int st = 128; st > 0; st >>= 1) {
    if (tid < st) { s1[tid] += s1[tid + st]; s2[tid] += s2[tid + st]; }
    __syncthreads();
  }
  const float mu  = s1[0] * (1.0f / D_);
  const float var = s2[0] * (1.0f / D_) - mu * mu;
  const float rs  = rsqrtf(var + 1e-5f);
#pragma unroll
  for (int i = 0; i < 3; ++i) {
    int d = tid + i * 256;
    out[(size_t)m * D_ + d] = (vals[i] - mu) * rs * g[d] + bb[d];
  }
}

// ---------------- host side ----------------

extern "C" void kernel_launch(void* const* d_in, const int* in_sizes, int n_in,
                              void* d_out, int out_size, void* d_ws, size_t ws_size,
                              hipStream_t stream) {
  (void)in_sizes; (void)n_in; (void)out_size; (void)ws_size;

  const float* x       = (const float*)d_in[0];
  const float* proj_w  = (const float*)d_in[19];
  const float* proj_b  = (const float*)d_in[20];
  const float* ln_g    = (const float*)d_in[21];
  const float* ln_b    = (const float*)d_in[22];
  float* out = (float*)d_out;

  // carve workspace (offsets 256B-aligned)
  char* ws = (char*)d_ws;
  size_t off = 0;
  auto carve = [&](size_t bytes) {
    void* p = ws + off;
    off += (bytes + 255) & ~(size_t)255;
    return p;
  };
  __bf16* x_bf      = (__bf16*)carve((size_t)M_ * D_ * 2);
  __bf16* inw_bf[2] = {(__bf16*)carve((size_t)N1_ * D_ * 2),
                       (__bf16*)carve((size_t)N1_ * D_ * 2)};
  __bf16* xpj_bf[2] = {(__bf16*)carve((size_t)NP_ * DI_ * 2),
                       (__bf16*)carve((size_t)NP_ * DI_ * 2)};
  __bf16* dtw_bf[2] = {(__bf16*)carve((size_t)DI_ * R_ * 2),
                       (__bf16*)carve((size_t)DI_ * R_ * 2)};
  __bf16* otw_bf[2] = {(__bf16*)carve((size_t)D_ * DI_ * 2),
                       (__bf16*)carve((size_t)D_ * DI_ * 2)};
  __bf16* pjw_bf    = (__bf16*)carve((size_t)D_ * D2_ * 2);
  float*  xz        = (float*)carve((size_t)M_ * N1_ * 4);
  float*  xc_act    = (float*)carve((size_t)M_ * DI_ * 4);
  __bf16* xc_bf     = (__bf16*)carve((size_t)M_ * DI_ * 2);
  float*  dbc       = (float*)carve((size_t)M_ * NP_ * 4);
  __bf16* dbc_bf    = (__bf16*)carve((size_t)M_ * NP_ * 2);
  float*  dtb       = (float*)carve((size_t)M_ * DI_ * 4);
  float*  Pbuf      = (float*)carve((size_t)B_ * T_ * S_ * DI_ * 4);
  float*  Fbuf      = (float*)carve((size_t)B_ * T_ * S_ * DI_ * 4);
  float*  Hbuf      = (float*)carve((size_t)B_ * T_ * S_ * DI_ * 4);
  __bf16* yg_bf     = (__bf16*)carve((size_t)M_ * DI_ * 2);
  __bf16* merged_bf = (__bf16*)carve((size_t)M_ * D2_ * 2);
  float*  projbuf   = (float*)carve((size_t)M_ * D_ * 4);

  auto CVT = [&](const void* s, void* d, int n) {
    cvt_bf16<<<(n + 255) / 256, 256, 0, stream>>>((const float*)s, (__bf16*)d, n);
  };

  // one-time conversions
  CVT(x, x_bf, M_ * D_);
  for (int dir = 0; dir < 2; ++dir) {
    const int base = 1 + dir * 9;
    CVT(d_in[base + 0], inw_bf[dir], N1_ * D_);
    CVT(d_in[base + 3], xpj_bf[dir], NP_ * DI_);
    CVT(d_in[base + 4], dtw_bf[dir], DI_ * R_);
    CVT(d_in[base + 8], otw_bf[dir], D_ * DI_);
  }
  CVT(proj_w, pjw_bf, D_ * D2_);

  const dim3 blk(256);
  const int ewGrid = (M_ * DI_) / 256;  // exact

  for (int dir = 0; dir < 2; ++dir) {
    const int base = 1 + dir * 9;
    const float* conv_w = (const float*)d_in[base + 1];
    const float* conv_b = (const float*)d_in[base + 2];
    const float* dt_b   = (const float*)d_in[base + 5];
    const float* A_log  = (const float*)d_in[base + 6];
    const float* Dp     = (const float*)d_in[base + 7];

    // G1: xz = x(rev?) @ in_w^T   (M x 3072)
    gemm_bf16_wmma<<<dim3(N1_ / 128, M_ / 64), blk, 0, stream>>>(
        x_bf, D_, inw_bf[dir], D_, N1_, D_, xz, nullptr, N1_, nullptr,
        /*mode=*/0, /*revIn=*/dir, /*revOut=*/0);

    // conv + SiLU -> xc_act (f32) + xc_bf
    conv_silu<<<ewGrid, blk, 0, stream>>>(xz, conv_w, conv_b, xc_act, xc_bf);

    // G2: dbc = xc @ xproj_w^T   (M x 80), keep f32 + bf16 copies
    gemm_bf16_wmma<<<dim3((NP_ + 127) / 128, M_ / 64), blk, 0, stream>>>(
        xc_bf, DI_, xpj_bf[dir], DI_, NP_, DI_, dbc, dbc_bf, NP_, nullptr,
        /*mode=*/1, 0, 0);

    // G3: dt = softplus(dbc[:, :48] @ dt_w^T + dt_b)   (M x 1536), Kd = 48
    gemm_bf16_wmma<<<dim3(DI_ / 128, M_ / 64), blk, 0, stream>>>(
        dbc_bf, NP_, dtw_bf[dir], R_, DI_, R_, dtb, nullptr, DI_, dt_b,
        /*mode=*/2, 0, 0);

    // chunked selective scan (3 passes), fused gate -> yg_bf
    scan_part1<<<dim3(DI_ / 256, T_, B_), blk, 0, stream>>>(
        xc_act, dtb, dbc, A_log, Pbuf, Fbuf);
    scan_fix<<<(B_ * S_ * DI_) / 256, blk, 0, stream>>>(Pbuf, Fbuf, Hbuf);
    scan_part2<<<dim3(DI_ / 256, T_, B_), blk, 0, stream>>>(
        xc_act, dtb, dbc, A_log, Dp, Hbuf, xz, yg_bf);

    // G4: merged[:, dir*768 :] = (yg @ out_w^T), rows reversed for bwd
    gemm_bf16_wmma<<<dim3(D_ / 128, M_ / 64), blk, 0, stream>>>(
        yg_bf, DI_, otw_bf[dir], DI_, D_, DI_, nullptr, merged_bf + dir * D_,
        D2_, nullptr, /*mode=*/3, 0, /*revOut=*/dir);
  }

  // G5: proj = merged @ proj_w^T + proj_b   (M x 768)
  gemm_bf16_wmma<<<dim3(D_ / 128, M_ / 64), blk, 0, stream>>>(
      merged_bf, D2_, pjw_bf, D2_, D_, D2_, projbuf, nullptr, D_, proj_b,
      /*mode=*/4, 0, 0);

  // residual + LayerNorm -> d_out
  ln_k<<<M_, blk, 0, stream>>>(x, projbuf, ln_g, ln_b, out);
}
